// TemporalDecoder_20718922236134
// MI455X (gfx1250) — matrix operand order repaired and checked by reading, not compiled, not run on hardware
//
#include <hip/hip_runtime.h>

// ---------------------------------------------------------------------------
// TemporalDecoder for MI455X (gfx1250, wave32, WMMA bf16 16x16x32, f32 accum)
// B=32 S=64 N=100 H=256 F=22 steps=10 ; LSTM batch = B*N = 3200 rows.
// ---------------------------------------------------------------------------

typedef __attribute__((ext_vector_type(16))) __bf16 v16bf;
typedef __attribute__((ext_vector_type(8)))  float  v8f;

#define B_   32
#define S_   64
#define N_   100
#define H_   256
#define F_   22
#define BN_  3200

__device__ __forceinline__ float sigf(float x) { return 1.0f / (1.0f + expf(-x)); }

__device__ __forceinline__ v8f wmma_bf16(v16bf a, v16bf b, v8f c) {
  // D = A(16x32 bf16) x B(32x16 bf16) + C(16x16 f32)
  return __builtin_amdgcn_wmma_f32_16x16x32_bf16(false, a, false, b, (short)0, c,
                                                 false, false);
}

// A fragment from an LDS tile laid out [16 rows][256 cols] bf16, row-major.
// ISA 16-bit A 16x32 layout: lane m=l&15; lanes<16 hold K 0..7 & 16..23 of the
// k-slice, lanes>=16 hold K 8..15 & 24..31.
__device__ __forceinline__ v16bf load_a_frag(const __bf16* a, int tk, int lane) {
  const int m  = lane & 15;
  const int kb = tk * 32 + ((lane < 16) ? 0 : 8);
  const __bf16* p0 = a + m * H_ + kb;
  v16bf f;
#pragma unroll
  for (int j = 0; j < 8; ++j) { f[j] = p0[j]; f[j + 8] = p0[16 + j]; }
  return f;
}

// B fragments are pre-packed so each lane's 16 bf16 are contiguous (32B).
__device__ __forceinline__ v16bf load_b_frag(const __bf16* packed, int tk, int tn,
                                             int NT, int lane) {
  return *(reinterpret_cast<const v16bf*>(packed) +
           (size_t)((tk * NT + tn) * 32 + lane));
}

// Stage 16 rows x 256 cols of f32 -> bf16 LDS tile. 256 threads, 16 elems each.
__device__ __forceinline__ void stage_tile(__bf16* dst, const float* src, int R0,
                                           int tid) {
#pragma unroll
  for (int e = 0; e < 16; ++e)
    dst[e * 256 + tid] = (__bf16)src[(size_t)(R0 + e) * H_ + tid];
}

// Gather x[b, t, n, :] for rows r = b*N + n (x is [B,S,N,H]).
__device__ __forceinline__ void stage_tile_x(__bf16* dst, const float* x, int R0,
                                             int t, int tid) {
#pragma unroll
  for (int e = 0; e < 16; ++e) {
    const int gr = R0 + e;
    const int bb = gr / N_, nn = gr % N_;
    dst[e * 256 + tid] =
        (__bf16)x[(((size_t)bb * S_ + t) * N_ + nn) * H_ + tid];
  }
}

// ---------------------------------------------------------------------------
// Pack W (row-major [Nrows, K]) into WMMA-B fragment order, bf16, transposed:
// B[k][n] = W[n][k]; tiles (tk: K/32, tn: Ndst/16), 32 lanes x 16 elems each.
// ---------------------------------------------------------------------------
__global__ void pack_b_kernel(const float* __restrict__ W, __bf16* __restrict__ dst,
                              int K, int Nsrc, int Ndst) {
  const int total = (K / 32) * (Ndst / 16) * 512;
  const int p = blockIdx.x * blockDim.x + threadIdx.x;
  if (p >= total) return;
  const int j    = p & 15;
  const int l    = (p >> 4) & 31;
  const int tile = p >> 9;
  const int NT   = Ndst / 16;
  const int tn   = tile % NT;
  const int tk   = tile / NT;
  const int k    = tk * 32 + ((l < 16) ? 0 : 8) + (j & 7) + ((j < 8) ? 0 : 16);
  const int n    = tn * 16 + (l & 15);
  const float v  = (n < Nsrc) ? W[(size_t)n * K + k] : 0.0f;
  dst[p] = (__bf16)v;
}

__global__ void zero_kernel(float* __restrict__ p, int n) {
  const int i = blockIdx.x * blockDim.x + threadIdx.x;
  if (i < n) p[i] = 0.0f;
}

// ---------------------------------------------------------------------------
// One LSTM timestep, BOTH layers fused. 256 threads = 8 waves, 16 rows per WG.
// mode 0: input gathered from x at time t; mode 1: input is plain [BN,H] (h_t).
// ---------------------------------------------------------------------------
__global__ __launch_bounds__(256) void lstm_step_kernel(
    const float* __restrict__ xin, int mode, int t,
    const __bf16* __restrict__ Wih0p, const __bf16* __restrict__ Whh0p,
    const float* __restrict__ b0,
    const __bf16* __restrict__ Wih1p, const __bf16* __restrict__ Whh1p,
    const float* __restrict__ b1,
    float* __restrict__ h0, float* __restrict__ c0,
    float* __restrict__ h1, float* __restrict__ c1) {
  __shared__ __bf16 sIn[16 * 256];
  __shared__ __bf16 sH0p[16 * 256];
  __shared__ __bf16 sH1p[16 * 256];
  __shared__ __bf16 sH0n[16 * 256];
  __shared__ float  sG[16 * 1024];

  const int tid  = threadIdx.x;
  const int lane = tid & 31;
  const int w    = tid >> 5;
  const int R0   = blockIdx.x * 16;

  if (mode == 0) stage_tile_x(sIn, xin, R0, t, tid);
  else           stage_tile(sIn, xin, R0, tid);
  stage_tile(sH0p, h0, R0, tid);
  stage_tile(sH1p, h1, R0, tid);
  __syncthreads();

  // ---------- layer 0: gates = x@Wih0^T + h0@Whh0^T ----------
  {
    v8f z = {};
    v8f acc[8];
#pragma unroll
    for (int tn = 0; tn < 8; ++tn) acc[tn] = z;
#pragma unroll
    for (int tk = 0; tk < 8; ++tk) {
      const v16bf aX = load_a_frag(sIn, tk, lane);
      const v16bf aH = load_a_frag(sH0p, tk, lane);
#pragma unroll
      for (int tn = 0; tn < 8; ++tn) {
        acc[tn] = wmma_bf16(aX, load_b_frag(Wih0p, tk, w * 8 + tn, 64, lane), acc[tn]);
        acc[tn] = wmma_bf16(aH, load_b_frag(Whh0p, tk, w * 8 + tn, 64, lane), acc[tn]);
      }
    }
    const int rb   = (lane < 16) ? 0 : 8;
    const int colb = w * 128 + (lane & 15);
#pragma unroll
    for (int tn = 0; tn < 8; ++tn)
#pragma unroll
      for (int r8 = 0; r8 < 8; ++r8)
        sG[(rb + r8) * 1024 + colb + tn * 16] = acc[tn][r8];
  }
  __syncthreads();

  // ---------- layer 0 elementwise ----------
  {
    const int j = tid;
    const float bi = b0[j], bf = b0[256 + j], bg = b0[512 + j], bo = b0[768 + j];
#pragma unroll 4
    for (int r = 0; r < 16; ++r) {
      const size_t g = (size_t)(R0 + r) * H_ + j;
      const float gi = sG[r * 1024 + j] + bi;
      const float gf = sG[r * 1024 + 256 + j] + bf;
      const float gg = sG[r * 1024 + 512 + j] + bg;
      const float go = sG[r * 1024 + 768 + j] + bo;
      const float cn = sigf(gf) * c0[g] + sigf(gi) * tanhf(gg);
      const float hn = sigf(go) * tanhf(cn);
      c0[g] = cn;
      h0[g] = hn;
      sH0n[r * 256 + j] = (__bf16)hn;   // layer-1 input tile
    }
  }
  __syncthreads();

  // ---------- layer 1: gates = h0new@Wih1^T + h1@Whh1^T ----------
  {
    v8f z = {};
    v8f acc[8];
#pragma unroll
    for (int tn = 0; tn < 8; ++tn) acc[tn] = z;
#pragma unroll
    for (int tk = 0; tk < 8; ++tk) {
      const v16bf aX = load_a_frag(sH0n, tk, lane);
      const v16bf aH = load_a_frag(sH1p, tk, lane);
#pragma unroll
      for (int tn = 0; tn < 8; ++tn) {
        acc[tn] = wmma_bf16(aX, load_b_frag(Wih1p, tk, w * 8 + tn, 64, lane), acc[tn]);
        acc[tn] = wmma_bf16(aH, load_b_frag(Whh1p, tk, w * 8 + tn, 64, lane), acc[tn]);
      }
    }
    const int rb   = (lane < 16) ? 0 : 8;
    const int colb = w * 128 + (lane & 15);
#pragma unroll
    for (int tn = 0; tn < 8; ++tn)
#pragma unroll
      for (int r8 = 0; r8 < 8; ++r8)
        sG[(rb + r8) * 1024 + colb + tn * 16] = acc[tn][r8];
  }
  __syncthreads();

  // ---------- layer 1 elementwise ----------
  {
    const int j = tid;
    const float bi = b1[j], bf = b1[256 + j], bg = b1[512 + j], bo = b1[768 + j];
#pragma unroll 4
    for (int r = 0; r < 16; ++r) {
      const size_t g = (size_t)(R0 + r) * H_ + j;
      const float gi = sG[r * 1024 + j] + bi;
      const float gf = sG[r * 1024 + 256 + j] + bf;
      const float gg = sG[r * 1024 + 512 + j] + bg;
      const float go = sG[r * 1024 + 768 + j] + bo;
      const float cn = sigf(gf) * c1[g] + sigf(gi) * tanhf(gg);
      const float hn = sigf(go) * tanhf(cn);
      c1[g] = cn;
      h1[g] = hn;
    }
  }
}

// ---------------------------------------------------------------------------
// LayerNorm over H=256: one block per row.
// ---------------------------------------------------------------------------
__global__ __launch_bounds__(256) void ln_kernel(const float* __restrict__ src,
                                                 const float* __restrict__ g,
                                                 const float* __restrict__ b,
                                                 float* __restrict__ dst) {
  __shared__ float r1[256];
  __shared__ float r2[256];
  const int row = blockIdx.x, tid = threadIdx.x;
  const float v = src[(size_t)row * H_ + tid];
  r1[tid] = v;
  r2[tid] = v * v;
  __syncthreads();
  for (int s = 128; s > 0; s >>= 1) {
    if (tid < s) { r1[tid] += r1[tid + s]; r2[tid] += r2[tid + s]; }
    __syncthreads();
  }
  const float mu  = r1[0] * (1.0f / 256.0f);
  const float var = r2[0] * (1.0f / 256.0f) - mu * mu;
  dst[(size_t)row * H_ + tid] = (v - mu) * rsqrtf(var + 1e-5f) * g[tid] + b[tid];
}

// ---------------------------------------------------------------------------
// pred = relu(ht@fc1^T + b) @ fc2^T + b2, scattered to out[b, s, n, f].
// fc2 packed with N padded 22 -> 32. 16 rows per WG.
// ---------------------------------------------------------------------------
__global__ __launch_bounds__(256) void pred_kernel(
    const float* __restrict__ ht,
    const __bf16* __restrict__ fc1p, const float* __restrict__ fc1b,
    const __bf16* __restrict__ fc2p, const float* __restrict__ fc2b,
    float* __restrict__ out, int s, int steps) {
  __shared__ __bf16 sIn[16 * 256];
  __shared__ __bf16 sMid[16 * 256];
  const int tid = threadIdx.x, lane = tid & 31, w = tid >> 5;
  const int R0 = blockIdx.x * 16;

  stage_tile(sIn, ht, R0, tid);
  __syncthreads();

  {  // fc1 + relu -> sMid (bf16 A-tile for fc2)
    v8f z = {};
    v8f acc[2];
    acc[0] = z; acc[1] = z;
#pragma unroll
    for (int tk = 0; tk < 8; ++tk) {
      const v16bf a = load_a_frag(sIn, tk, lane);
#pragma unroll
      for (int tn = 0; tn < 2; ++tn)
        acc[tn] = wmma_bf16(a, load_b_frag(fc1p, tk, w * 2 + tn, 16, lane), acc[tn]);
    }
    const int rb = (lane < 16) ? 0 : 8;
#pragma unroll
    for (int tn = 0; tn < 2; ++tn) {
      const int col = (w * 2 + tn) * 16 + (lane & 15);
      const float bb = fc1b[col];
#pragma unroll
      for (int r8 = 0; r8 < 8; ++r8)
        sMid[(rb + r8) * 256 + col] = (__bf16)fmaxf(acc[tn][r8] + bb, 0.0f);
    }
  }
  __syncthreads();

  if (w < 2) {  // fc2 (N=32 padded), waves 0..1 only (wave-uniform branch)
    v8f acc = {};
#pragma unroll
    for (int tk = 0; tk < 8; ++tk) {
      const v16bf a = load_a_frag(sMid, tk, lane);
      acc = wmma_bf16(a, load_b_frag(fc2p, tk, w, 2, lane), acc);
    }
    const int col = w * 16 + (lane & 15);
    if (col < F_) {
      const int rb = (lane < 16) ? 0 : 8;
      const float bb = fc2b[col];
#pragma unroll
      for (int r8 = 0; r8 < 8; ++r8) {
        const int gr = R0 + rb + r8;
        const int bI = gr / N_, nn = gr % N_;
        out[(((size_t)bI * steps + s) * N_ + nn) * F_ + col] = acc[r8] + bb;
      }
    }
  }
}

// ---------------------------------------------------------------------------
extern "C" void kernel_launch(void* const* d_in, const int* in_sizes, int n_in,
                              void* d_out, int out_size, void* d_ws, size_t ws_size,
                              hipStream_t stream) {
  const float* x     = (const float*)d_in[0];
  const float* Wih0  = (const float*)d_in[2];
  const float* Whh0  = (const float*)d_in[3];
  const float* b0    = (const float*)d_in[4];
  const float* Wih1  = (const float*)d_in[5];
  const float* Whh1  = (const float*)d_in[6];
  const float* b1    = (const float*)d_in[7];
  const float* lng   = (const float*)d_in[8];
  const float* lnb   = (const float*)d_in[9];
  const float* fc1W  = (const float*)d_in[10];
  const float* fc1b  = (const float*)d_in[11];
  const float* fc2W  = (const float*)d_in[12];
  const float* fc2b  = (const float*)d_in[13];
  float* out = (float*)d_out;

  int steps = out_size / (B_ * N_ * F_);
  if (steps <= 0) steps = 1;

  // ---- workspace layout ----
  char* ws = (char*)d_ws;
  const size_t WBIG = 256 * 1024;  // elements in one packed 256x1024 matrix
  __bf16* Wih0p = (__bf16*)ws;
  __bf16* Whh0p = Wih0p + WBIG;
  __bf16* Wih1p = Whh0p + WBIG;
  __bf16* Whh1p = Wih1p + WBIG;
  __bf16* fc1p  = Whh1p + WBIG;            // 256*256 elements
  __bf16* fc2p  = fc1p + 256 * 256;        // 256*32 elements
  float* fbase  = (float*)(ws + 2244608);  // 32B-aligned
  const size_t RS = (size_t)BN_ * H_;      // 819200
  float* h0 = fbase;
  float* c0 = h0 + RS;
  float* h1 = c0 + RS;
  float* c1 = h1 + RS;
  float* ht = c1 + RS;

  // ---- pack weights (bf16, transposed, WMMA-B fragment order) ----
  pack_b_kernel<<<1024, 256, 0, stream>>>(Wih0, Wih0p, 256, 1024, 1024);
  pack_b_kernel<<<1024, 256, 0, stream>>>(Whh0, Whh0p, 256, 1024, 1024);
  pack_b_kernel<<<1024, 256, 0, stream>>>(Wih1, Wih1p, 256, 1024, 1024);
  pack_b_kernel<<<1024, 256, 0, stream>>>(Whh1, Whh1p, 256, 1024, 1024);
  pack_b_kernel<<<256, 256, 0, stream>>>(fc1W, fc1p, 256, 256, 256);
  pack_b_kernel<<<32, 256, 0, stream>>>(fc2W, fc2p, 256, F_, 32);

  // ---- zero h/c state (deterministic every call) ----
  zero_kernel<<<(int)((4 * RS + 255) / 256), 256, 0, stream>>>(h0, (int)(4 * RS));

  // ---- sequence phase: 64 fused two-layer LSTM steps ----
  for (int t = 0; t < S_; ++t)
    lstm_step_kernel<<<BN_ / 16, 256, 0, stream>>>(x, 0, t, Wih0p, Whh0p, b0,
                                                   Wih1p, Whh1p, b1, h0, c0, h1, c1);

  // h_t = LN(h1 at t=63)
  ln_kernel<<<BN_, 256, 0, stream>>>(h1, lng, lnb, ht);

  // ---- autoregressive phase ----
  for (int s = 0; s < steps; ++s) {
    pred_kernel<<<BN_ / 16, 256, 0, stream>>>(ht, fc1p, fc1b, fc2p, fc2b, out, s, steps);
    lstm_step_kernel<<<BN_ / 16, 256, 0, stream>>>(ht, 1, 0, Wih0p, Whh0p, b0,
                                                   Wih1p, Whh1p, b1, h0, c0, h1, c1);
    ln_kernel<<<BN_, 256, 0, stream>>>(h1, lng, lnb, ht);
  }
}
